// TimeMix_39530878992406
// MI455X (gfx1250) — matrix operand order repaired and checked by reading, not compiled, not run on hardware
//
#include <hip/hip_runtime.h>
#include <hip/hip_bf16.h>
#include <cstdint>

// ---------------------------------------------------------------------------
// RWKV-6 TimeMix for MI455X (gfx1250, wave32, WMMA).
// B=8, T=2048, C=1024, H=16, HS=64, LORA_R=32 (x5), TD_R=64.
// All dense GEMMs: v_wmma_f32_16x16x32_bf16, 128x128 block tile, 8 waves,
// each wave = 2x4 WMMA tiles. All B operands pre-transposed to [N][K] bf16
// so staging is always vectorized 16B chunks. Global->LDS staging uses the
// CDNA5 async-to-LDS path (global_load_async_to_lds_b128 / ASYNCcnt).
// N-guards are compile-time-selected so the hot N=1024 GEMMs stage branchless.
// ---------------------------------------------------------------------------

typedef __bf16 bf16_t;
typedef __attribute__((ext_vector_type(16))) __bf16 v16bf;
typedef __attribute__((ext_vector_type(8)))  float  v8f;
typedef __attribute__((ext_vector_type(4)))  unsigned int u32x4;
typedef __attribute__((ext_vector_type(4)))  int    i32x4;
typedef __attribute__((ext_vector_type(4)))  float  f32x4;

#define GLOBAL_AS __attribute__((address_space(1)))
#define LDS_AS    __attribute__((address_space(3)))

#if defined(__has_builtin)
#if __has_builtin(__builtin_amdgcn_global_load_async_to_lds_b128) && \
    __has_builtin(__builtin_amdgcn_s_wait_asynccnt)
#define HAVE_ASYNC_LDS 1
#endif
#endif

#define CB   1024
#define TT   2048
#define BB   8
#define HH   16
#define HSZ  64
#define BT   16384            // B*T
#define R5   160              // 5*LORA_R
#define TDR  64

// ----------------------------- f32 -> bf16 ---------------------------------
__global__ void cvt_bf16_kernel(const float* __restrict__ s,
                                bf16_t* __restrict__ d, int n) {
  int i = blockIdx.x * blockDim.x + threadIdx.x;
  if (i < n) d[i] = (bf16_t)s[i];
}

// -------------------- f32 [K][N] -> bf16 [N][K] transpose ------------------
__global__ void cvt_bf16_t_kernel(const float* __restrict__ s,
                                  bf16_t* __restrict__ d, int K, int N) {
  int i = blockIdx.x * blockDim.x + threadIdx.x;
  if (i < K * N) {
    int k = i / N, n = i % N;
    d[(size_t)n * K + k] = (bf16_t)s[i];
  }
}

// --------------------------- prep: dxa, xm ---------------------------------
__global__ void prep_kernel(const float* __restrict__ x,
                            const float* __restrict__ miu_x,
                            float* __restrict__ dxa,
                            bf16_t* __restrict__ xm) {
  long long idx = (long long)blockIdx.x * blockDim.x + threadIdx.x;
  int c = (int)(idx & (CB - 1));
  int t = (int)((idx >> 10) & (TT - 1));
  float xv = x[idx];
  float pv = (t > 0) ? x[idx - CB] : 0.f;
  float d = pv - xv;
  dxa[idx] = d;
  xm[idx] = (bf16_t)(xv + d * miu_x[c]);
}

// --------------------------- WMMA tiled GEMM -------------------------------
// C[M,N] = A[M,K](bf16 row-major, lda) @ B[N,K]^T (bf16 row-major, ldb).
// Block tile 128(M) x 128(N), K-step 32, 256 threads = 8 waves.
// Wave (wm in 0..3, wn in 0..1) owns a 32(M) x 64(N) region = 2x4 WMMA tiles.
// M must be a multiple of 128. GUARD_N=false requires N % 128 == 0.
// EPI: 0 f32 | 1 silu f32 | 2 tanh bf16 | 3 xx=x+dxa*(lam[n]+acc) bf16
//      4 e0[n]+acc f32
template <int EPI, bool GUARD_N>
__global__ __launch_bounds__(256)
void gemm_bf16_wmma(const bf16_t* __restrict__ Ag, int lda,
                    const bf16_t* __restrict__ Bg, int ldb,
                    int M, int N, int K,
                    void* __restrict__ Cout, int ldc,
                    const float* __restrict__ e0,
                    const float* __restrict__ e1,
                    const float* __restrict__ e2) {
  __shared__ bf16_t As[128][40];   // 32 K + pad (80B rows, 16B aligned)
  __shared__ bf16_t Bs[128][40];   // [n][k]

  const int tid  = threadIdx.x;
  const int lane = tid & 31;
  const int wid  = tid >> 5;
  const int wm   = wid & 3;        // M offset wm*32
  const int wn   = wid >> 2;       // N offset wn*64
  const int m0   = blockIdx.y * 128;
  const int n0   = blockIdx.x * 128;

  v8f acc[2][4] = {};

  const int chunk = tid & 3;       // 4 x 16B chunks per 32-K row

  for (int ks = 0; ks < K; ks += 32) {
    __syncthreads();               // previous iteration's frag reads done
#if HAVE_ASYNC_LDS
    #pragma unroll
    for (int it = 0; it < 2; ++it) {
      int row = it * 64 + (tid >> 2);
      __builtin_amdgcn_global_load_async_to_lds_b128(
          (GLOBAL_AS i32x4*)(Ag + (size_t)(m0 + row) * lda + ks + chunk * 8),
          (LDS_AS i32x4*)&As[row][chunk * 8], 0, 0);
      if (!GUARD_N || (n0 + row < N)) {
        __builtin_amdgcn_global_load_async_to_lds_b128(
            (GLOBAL_AS i32x4*)(Bg + (size_t)(n0 + row) * ldb + ks + chunk * 8),
            (LDS_AS i32x4*)&Bs[row][chunk * 8], 0, 0);
      } else {
        u32x4 z = {0u, 0u, 0u, 0u};
        *(u32x4*)&Bs[row][chunk * 8] = z;
      }
    }
    __builtin_amdgcn_s_wait_asynccnt(0);
#else
    #pragma unroll
    for (int it = 0; it < 2; ++it) {
      int row = it * 64 + (tid >> 2);
      *(u32x4*)&As[row][chunk * 8] =
          *(const u32x4*)(Ag + (size_t)(m0 + row) * lda + ks + chunk * 8);
      u32x4 bv = {0u, 0u, 0u, 0u};
      if (!GUARD_N || (n0 + row < N))
        bv = *(const u32x4*)(Bg + (size_t)(n0 + row) * ldb + ks + chunk * 8);
      *(u32x4*)&Bs[row][chunk * 8] = bv;
    }
#endif
    // prefetch next K tile (global_prefetch_b8)
    if (ks + 32 < K) {
      __builtin_prefetch(Ag + (size_t)(m0 + (tid >> 1)) * lda + ks + 32, 0, 0);
      __builtin_prefetch(Bg + (size_t)(n0 + (tid >> 1)) * ldb + ks + 32, 0, 0);
    }
    __syncthreads();

    // ---- fragments (documented wave32 16-bit WMMA layouts) ----
    // A 16x32: lane L -> M=L%16; VGPR0..3: K=(L/16)*8..+8, VGPR4..7: +16
    v16bf afr[2];
    #pragma unroll
    for (int ta = 0; ta < 2; ++ta) {
      int row  = wm * 32 + ta * 16 + (lane & 15);
      int half = lane >> 4;
      union { v16bf v; u32x4 q[2]; } f;
      const bf16_t* b = &As[row][0];
      f.q[0] = *(const u32x4*)(b + half * 8);
      f.q[1] = *(const u32x4*)(b + 16 + half * 8);
      afr[ta] = f.v;
    }
    // B 32x16: lane L -> N=L%16; VGPR0..3: K=(L/16)*8..+8, VGPR4..7: +16
    v16bf bfr[4];
    #pragma unroll
    for (int tb = 0; tb < 4; ++tb) {
      int col = wn * 64 + tb * 16 + (lane & 15);
      int kh  = lane >> 4;
      union { v16bf v; u32x4 q[2]; } f;
      const bf16_t* b = &Bs[col][0];
      f.q[0] = *(const u32x4*)(b + kh * 8);
      f.q[1] = *(const u32x4*)(b + 16 + kh * 8);
      bfr[tb] = f.v;
    }
    // ---- 8 WMMAs per wave per K-step ----
    #pragma unroll
    for (int ta = 0; ta < 2; ++ta)
      #pragma unroll
      for (int tb = 0; tb < 4; ++tb)
        acc[ta][tb] = __builtin_amdgcn_wmma_f32_16x16x32_bf16(
            false, afr[ta], false, bfr[tb], (short)0, acc[ta][tb], false, false);
  }

  // ---- epilogue + store (D layout: lane%16 = n, (lane/16)*8+vgpr = m) ----
  #pragma unroll
  for (int ta = 0; ta < 2; ++ta) {
    #pragma unroll
    for (int tb = 0; tb < 4; ++tb) {
      #pragma unroll
      for (int vg = 0; vg < 8; ++vg) {
        int m = m0 + wm * 32 + ta * 16 + (lane >> 4) * 8 + vg;
        int n = n0 + wn * 64 + tb * 16 + (lane & 15);
        if (GUARD_N && n >= N) continue;
        float val = acc[ta][tb][vg];
        size_t off = (size_t)m * ldc + n;
        if constexpr (EPI == 0) {
          ((float*)Cout)[off] = val;
        } else if constexpr (EPI == 1) {
          ((float*)Cout)[off] = val / (1.f + __expf(-val));        // silu
        } else if constexpr (EPI == 2) {
          ((bf16_t*)Cout)[off] = (bf16_t)tanhf(val);
        } else if constexpr (EPI == 3) {
          float mu = e0[n] + val;                                  // lambda + lora
          ((bf16_t*)Cout)[off] = (bf16_t)(e1[off] + e2[off] * mu); // x + dxa*mu
        } else if constexpr (EPI == 4) {
          ((float*)Cout)[off] = e0[n] + val;                       // td_miu + acc
        }
      }
    }
  }
}

// ------------------------------ WKV-6 scan ---------------------------------
// One block per (b,h). 256 threads: tid = j*4 + ic ; j = output dim (0..63),
// ic = 16-wide chunk of the k/state dim. S[16] lives in VGPRs across t.
__global__ __launch_bounds__(256, 2)
void wkv6_scan_kernel(const float* __restrict__ rB, const float* __restrict__ kB,
                      const float* __restrict__ vB, const float* __restrict__ wB,
                      const float* __restrict__ u,  float* __restrict__ yB) {
  const int bh = blockIdx.x;
  const int b  = bh / HH;
  const int h  = bh % HH;
  const int tid = threadIdx.x;
  const int j  = tid >> 2;
  const int ic = tid & 3;

  float S[16];
  float uloc[16];
  #pragma unroll
  for (int ii = 0; ii < 16; ++ii) {
    S[ii] = 0.f;
    uloc[ii] = u[h * HSZ + ic * 16 + ii];
  }

  __shared__ float sr[HSZ], sk[HSZ], sv[HSZ], sd[HSZ];

  for (int t = 0; t < TT; ++t) {
    size_t base = ((size_t)(b * TT + t)) * CB + h * HSZ;
    __syncthreads();
    if (tid < 64)        sr[tid]        = rB[base + tid];
    else if (tid < 128)  sk[tid - 64]   = kB[base + tid - 64];
    else if (tid < 192)  sv[tid - 128]  = vB[base + tid - 128];
    else { int i = tid - 192; sd[i] = __expf(-__expf(wB[base + i])); }
    __syncthreads();

    float vj = sv[j];
    float partial = 0.f;
    #pragma unroll
    for (int ii = 0; ii < 16; ++ii) {
      int i = ic * 16 + ii;
      float kv = sk[i] * vj;
      partial += sr[i] * (S[ii] + uloc[ii] * kv);
      S[ii] = sd[i] * S[ii] + kv;
    }
    partial += __shfl_xor(partial, 1, 32);
    partial += __shfl_xor(partial, 2, 32);
    if (ic == 0) yB[base + j] = partial;
  }
}

// ----------------------- GroupNorm(head) * g -> bf16 -----------------------
__global__ __launch_bounds__(256, 4)
void gn_mul_kernel(const float* __restrict__ y, const float* __restrict__ g,
                   const float* __restrict__ gamma, const float* __restrict__ beta,
                   bf16_t* __restrict__ out) {
  const int row = blockIdx.x;
  const int tid = threadIdx.x;
  const int c0  = tid * 4;
  const float* yr = y + (size_t)row * CB;
  f32x4 yv = *(const f32x4*)(yr + c0);

  float s  = yv.x + yv.y + yv.z + yv.w;
  float sq = yv.x*yv.x + yv.y*yv.y + yv.z*yv.z + yv.w*yv.w;
  #pragma unroll
  for (int m = 1; m < 16; m <<= 1) {    // 16 lanes = one head
    s  += __shfl_xor(s,  m, 16);
    sq += __shfl_xor(sq, m, 16);
  }
  float mean = s * (1.f / 64.f);
  float var  = sq * (1.f / 64.f) - mean * mean;
  float inv  = rsqrtf(var + 1.6e-4f);   // eps = 1e-5 * n_head

  const float* gr = g + (size_t)row * CB;
  f32x4 gv = *(const f32x4*)(gr + c0);
  #pragma unroll
  for (int q = 0; q < 4; ++q) {
    int c = c0 + q;
    float yy = (q == 0) ? yv.x : (q == 1) ? yv.y : (q == 2) ? yv.z : yv.w;
    float gg = (q == 0) ? gv.x : (q == 1) ? gv.y : (q == 2) ? gv.z : gv.w;
    float yn = (yy - mean) * inv * gamma[c] + beta[c];
    out[(size_t)row * CB + c] = (bf16_t)(yn * gg);
  }
}

// ------------------------------- host side ---------------------------------
extern "C" void kernel_launch(void* const* d_in, const int* in_sizes, int n_in,
                              void* d_out, int out_size, void* d_ws, size_t ws_size,
                              hipStream_t stream) {
  const float* x      = (const float*)d_in[0];
  const float* miu_x  = (const float*)d_in[1];
  const float* lam    = (const float*)d_in[2];   // (5,1,1,C)
  const float* A_lora = (const float*)d_in[3];   // (C,160)
  const float* B_lora = (const float*)d_in[4];   // (5,32,C)
  const float* td_miu = (const float*)d_in[5];
  const float* td_A   = (const float*)d_in[6];   // (C,64)
  const float* td_B   = (const float*)d_in[7];   // (64,C)
  const float* u      = (const float*)d_in[8];   // (16,64)
  const float* Wmats[5] = { (const float*)d_in[9],  (const float*)d_in[10],
                            (const float*)d_in[11], (const float*)d_in[12],
                            (const float*)d_in[13] };   // Wr,Wk,Wv,Wg,Wo
  const float* gamma  = (const float*)d_in[14];
  const float* beta   = (const float*)d_in[15];
  float* out = (float*)d_out;

  const size_t CC  = (size_t)CB * CB;
  const size_t BTC = (size_t)BT * CB;

  // workspace carve-up (256B aligned)
  char* base = (char*)d_ws;
  size_t o = 0;
  auto take = [&](size_t bytes) { char* p = base + o; o += (bytes + 255) & ~(size_t)255; return p; };
  bf16_t* wbf    = (bf16_t*)take(5 * CC * 2);               // Wr..Wo (bf16, [N][K] already)
  bf16_t* aloraT = (bf16_t*)take((size_t)R5 * CB * 2);      // (160,1024)
  bf16_t* bloraT = (bf16_t*)take((size_t)5 * CB * 32 * 2);  // 5 x (1024,32)
  bf16_t* tdAT   = (bf16_t*)take((size_t)TDR * CB * 2);     // (64,1024)
  bf16_t* tdBT   = (bf16_t*)take((size_t)CB * TDR * 2);     // (1024,64)
  bf16_t* xm     = (bf16_t*)take(BTC * 2);
  float*  dxa    = (float*) take(BTC * 4);
  bf16_t* lorat  = (bf16_t*)take((size_t)BT * R5 * 2);
  bf16_t* tdh    = (bf16_t*)take((size_t)BT * TDR * 2);
  bf16_t* xx     = (bf16_t*)take(5 * BTC * 2);              // [w,k,v,r,g]
  float*  rbuf   = (float*) take(BTC * 4);
  float*  kbuf   = (float*) take(BTC * 4);
  float*  vbuf   = (float*) take(BTC * 4);
  float*  wdec   = (float*) take(BTC * 4);
  float*  gbuf   = (float*) take(BTC * 4);
  float*  ybuf   = (float*) take(BTC * 4);
  bf16_t* yz     = (bf16_t*)take(BTC * 2);
  (void)ws_size; (void)in_sizes; (void)n_in; (void)out_size;

  auto cvt = [&](const float* s, bf16_t* d, size_t n) {
    cvt_bf16_kernel<<<dim3((unsigned)((n + 255) / 256)), 256, 0, stream>>>(s, d, (int)n);
  };
  auto cvtT = [&](const float* s, bf16_t* d, int K, int N) {
    size_t n = (size_t)K * N;
    cvt_bf16_t_kernel<<<dim3((unsigned)((n + 255) / 256)), 256, 0, stream>>>(s, d, K, N);
  };
  // 1) weights -> bf16; [K,N] mats pre-transposed to [N,K]
  for (int i = 0; i < 5; ++i) cvt(Wmats[i], wbf + i * CC, CC);  // W is (N,K) already
  cvtT(A_lora, aloraT, CB, R5);
  for (int f = 0; f < 5; ++f)
    cvtT(B_lora + (size_t)f * 32 * CB, bloraT + (size_t)f * CB * 32, 32, CB);
  cvtT(td_A, tdAT, CB, TDR);
  cvtT(td_B, tdBT, TDR, CB);

  // 2) prep: dxa, xm
  prep_kernel<<<dim3((unsigned)(BTC / 256)), 256, 0, stream>>>(x, miu_x, dxa, xm);

  auto grid2 = [](int M, int N) {
    return dim3((unsigned)((N + 127) / 128), (unsigned)((M + 127) / 128));
  };

  // 3) lora_h = tanh(xm @ A_lora)         (M=BT, N=160, K=1024) -> bf16
  gemm_bf16_wmma<2, true><<<grid2(BT, R5), 256, 0, stream>>>(
      xm, CB, aloraT, CB, BT, R5, CB, lorat, R5, nullptr, nullptr, nullptr);

  // 4) per-branch: xx_f = x + dxa*(lambda_f + lora_h_f @ B_lora_f)   (K=32)
  for (int f = 0; f < 5; ++f) {
    gemm_bf16_wmma<3, false><<<grid2(BT, CB), 256, 0, stream>>>(
        lorat + f * 32, R5, bloraT + (size_t)f * CB * 32, 32,
        BT, CB, 32, xx + (size_t)f * BTC, CB,
        lam + f * CB, x, dxa);
  }

  // 5) big GEMMs: r,k,v (f32), g (fused silu).   xx order: [w,k,v,r,g]
  gemm_bf16_wmma<0, false><<<grid2(BT, CB), 256, 0, stream>>>(
      xx + 3 * BTC, CB, wbf + 0 * CC, CB, BT, CB, CB, rbuf, CB, nullptr, nullptr, nullptr);
  gemm_bf16_wmma<0, false><<<grid2(BT, CB), 256, 0, stream>>>(
      xx + 1 * BTC, CB, wbf + 1 * CC, CB, BT, CB, CB, kbuf, CB, nullptr, nullptr, nullptr);
  gemm_bf16_wmma<0, false><<<grid2(BT, CB), 256, 0, stream>>>(
      xx + 2 * BTC, CB, wbf + 2 * CC, CB, BT, CB, CB, vbuf, CB, nullptr, nullptr, nullptr);
  gemm_bf16_wmma<1, false><<<grid2(BT, CB), 256, 0, stream>>>(
      xx + 4 * BTC, CB, wbf + 3 * CC, CB, BT, CB, CB, gbuf, CB, nullptr, nullptr, nullptr);

  // 6) time-decay chain: tdh = tanh(xx_w @ td_A); w = td_miu + tdh @ td_B
  gemm_bf16_wmma<2, true><<<grid2(BT, TDR), 256, 0, stream>>>(
      xx + 0 * BTC, CB, tdAT, CB, BT, TDR, CB, tdh, TDR, nullptr, nullptr, nullptr);
  gemm_bf16_wmma<4, false><<<grid2(BT, CB), 256, 0, stream>>>(
      tdh, TDR, tdBT, TDR, BT, CB, TDR, wdec, CB, td_miu, nullptr, nullptr);

  // 7) WKV-6 scan (128 chains)
  wkv6_scan_kernel<<<dim3(BB * HH), 256, 0, stream>>>(rbuf, kbuf, vbuf, wdec, u, ybuf);

  // 8) GroupNorm * g -> bf16
  gn_mul_kernel<<<dim3(BT), 256, 0, stream>>>(ybuf, gbuf, gamma, beta, yz);

  // 9) out = (yn*g) @ Wo^T  -> f32
  gemm_bf16_wmma<0, false><<<grid2(BT, CB), 256, 0, stream>>>(
      yz, CB, wbf + 4 * CC, CB, BT, CB, CB, out, CB, nullptr, nullptr, nullptr);
}
